// SomaMind_22290880266265
// MI455X (gfx1250) — matrix operand order repaired and checked by reading, not compiled, not run on hardware
//
#include <hip/hip_runtime.h>
#include <cstdint>
#include <cstddef>

typedef __bf16 bf16_t;
typedef __attribute__((ext_vector_type(16))) bf16_t v16bf;
typedef __attribute__((ext_vector_type(8)))  bf16_t v8bf;
typedef __attribute__((ext_vector_type(8)))  float  v8f;

#define NEGBIG (-1000000000.0f)
#define BB 64
#define LL 512
#define EE 256
#define HH 512
#define DD 1024
#define TT 8
#define VOCAB 32000
#define OPE_D 64
#define RD_D 128
#define START_OP 64
#define NBLK 8   // persistent LSTM grid size

__device__ __forceinline__ float sigf(float x) { return 1.0f / (1.0f + __expf(-x)); }

// ---------------- WMMA fragment loaders (contiguous k order, k = 16*(lane>>4) + s) ----------------
// A: 16x32 bf16. lanes 0-15: K {0..15}; lanes 16-31: K {16..31}.
__device__ __forceinline__ v16bf load_a_frag(const bf16_t* rowk, int lane) {
  return *(const v16bf*)(rowk + ((lane >> 4) << 4));
}
// B: 32x16 bf16. lane n holds col n&15, K = 16 contiguous values starting at (lane>>4)*16.
// colk points at W[col*ldw + kk] of the (N,K) row-major weight.
__device__ __forceinline__ v16bf load_b_frag(const bf16_t* colk, int lane) {
  return *(const v16bf*)(colk + ((lane >> 4) << 4));
}
__device__ __forceinline__ v8f wmma_bf16(v16bf a, v16bf b, v8f c) {
  return __builtin_amdgcn_wmma_f32_16x16x32_bf16(false, a, false, b, (short)0, c, false, false);
}

// ---------------- Generic GEMM: C(M,N) = A(M,K) x W(N,K)^T + bias ----------------
// block 256 thr = 8 waves, block tile 64(M) x 128(N), wave tile 32x32.
// M must be a multiple of 64 here (always 64). N arbitrary (clamped). K multiple of 32.
__global__ void sm_gemm_bf16(const bf16_t* __restrict__ A, int lda,
                             const bf16_t* __restrict__ W, int ldw,
                             const float* __restrict__ bias,
                             float* __restrict__ C, int ldc,
                             bf16_t* __restrict__ Cb, int ldcb,
                             int M, int N, int K, int act) {
  const int tid = threadIdx.x, lane = tid & 31, wave = tid >> 5;
  const int wm = wave & 1, wn = wave >> 1;
  const int bm = blockIdx.y * 64, bn = blockIdx.x * 128;

  v8f zero = {};
  v8f acc[2][2];
  acc[0][0] = zero; acc[0][1] = zero; acc[1][0] = zero; acc[1][1] = zero;

  const int r0 = bm + wm * 32 + (lane & 15);
  const bf16_t* ap0 = A + (size_t)r0 * lda;
  const bf16_t* ap1 = A + (size_t)(r0 + 16) * lda;
  int cidx[2]; const bf16_t* wp[2];
#pragma unroll
  for (int j = 0; j < 2; ++j) {
    int c = bn + wn * 32 + j * 16 + (lane & 15);
    cidx[j] = c;
    int cc = c < N ? c : (N - 1);
    wp[j] = W + (size_t)cc * ldw;
  }
  for (int kk = 0; kk < K; kk += 32) {
    __builtin_prefetch(wp[0] + kk + 128, 0, 1);   // global_prefetch_b8 into L2
    v16bf a0 = load_a_frag(ap0 + kk, lane);
    v16bf a1 = load_a_frag(ap1 + kk, lane);
    v16bf b0 = load_b_frag(wp[0] + kk, lane);
    v16bf b1 = load_b_frag(wp[1] + kk, lane);
    acc[0][0] = wmma_bf16(a0, b0, acc[0][0]);
    acc[0][1] = wmma_bf16(a0, b1, acc[0][1]);
    acc[1][0] = wmma_bf16(a1, b0, acc[1][0]);
    acc[1][1] = wmma_bf16(a1, b1, acc[1][1]);
  }
#pragma unroll
  for (int i = 0; i < 2; ++i)
#pragma unroll
    for (int j = 0; j < 2; ++j) {
      int c = cidx[j];
      if (c >= N) continue;
      float bb = bias ? bias[c] : 0.0f;
#pragma unroll
      for (int v = 0; v < 8; ++v) {
        int m = bm + wm * 32 + i * 16 + v + ((lane >> 4) << 3);
        if (m >= M) continue;
        float val = acc[i][j][v] + bb;
        if (act == 1) val = tanhf(val);
        if (C)  C[(size_t)m * ldc + c] = val;
        if (Cb) Cb[(size_t)m * ldcb + c] = (bf16_t)val;
      }
    }
}

// ---------------- Persistent BiLSTM direction kernel ----------------
__device__ __forceinline__ void grid_barrier(unsigned* cnt) {
  __syncthreads();
  if (threadIdx.x == 0) {
    __threadfence();
    unsigned old = atomicAdd(cnt, 1u);
    unsigned target = (old / NBLK + 1u) * NBLK;
    while (atomicAdd(cnt, 0u) < target) { __builtin_amdgcn_s_sleep(1); }
  }
  __syncthreads();
}

// Wcat: (2048, K=H+in_d) bf16, row r = [W_hh[r] | W_ih[r]]. bias = b_ih + b_hh.
// gates = [h_t | x_t] @ Wcat^T + bias, per-WG slice of 64 h-columns (256 gate rows).
__global__ void sm_lstm_persist(const bf16_t* __restrict__ Wcat,
                                const float* __restrict__ bias,
                                const bf16_t* __restrict__ x,     // (B,L,in_d)
                                const int* __restrict__ lengths,
                                bf16_t* __restrict__ hbuf,        // (B,H)
                                float* __restrict__ cbuf,         // (B,H)
                                bf16_t* __restrict__ enc,         // (B,L,2H)
                                unsigned* __restrict__ sync,
                                int in_d, int reverse, int dir) {
  __shared__ float gsm[256 * 64];   // [gate-row-local][batch]  (64KB)
  const int K = HH + in_d;
  const int tid = threadIdx.x, lane = tid & 31, wave = tid >> 5;
  const int wm = wave & 1;          // 32-batch half
  const int wn = wave >> 1;         // gate index 0..3 (i,f,g,o)
  const int slice = blockIdx.x * 64;

  const int b0 = wm * 32 + (lane & 15);
  const int b1 = b0 + 16;
  const bf16_t* hr0 = hbuf + (size_t)b0 * HH;
  const bf16_t* hr1 = hbuf + (size_t)b1 * HH;
  const bf16_t* wrow[4];
#pragma unroll
  for (int j = 0; j < 4; ++j) {
    int row = wn * HH + slice + j * 16 + (lane & 15);
    wrow[j] = Wcat + (size_t)row * K;
  }

  for (int step = 0; step < LL; ++step) {
    const int ts = reverse ? (LL - 1 - step) : step;
    v8f zero = {};
    v8f acc[2][4];
#pragma unroll
    for (int i = 0; i < 2; ++i)
#pragma unroll
      for (int j = 0; j < 4; ++j) acc[i][j] = zero;

    const bf16_t* xr0 = x + ((size_t)b0 * LL + ts) * in_d;
    const bf16_t* xr1 = x + ((size_t)b1 * LL + ts) * in_d;

    for (int kk = 0; kk < K; kk += 32) {
      const bf16_t* a0p = (kk < HH) ? (hr0 + kk) : (xr0 + (kk - HH));
      const bf16_t* a1p = (kk < HH) ? (hr1 + kk) : (xr1 + (kk - HH));
      v16bf a0 = load_a_frag(a0p, lane);
      v16bf a1 = load_a_frag(a1p, lane);
#pragma unroll
      for (int j = 0; j < 4; ++j) {
        v16bf bv = load_b_frag(wrow[j] + kk, lane);
        acc[0][j] = wmma_bf16(a0, bv, acc[0][j]);
        acc[1][j] = wmma_bf16(a1, bv, acc[1][j]);
      }
    }
    // dump gates (+bias) to LDS
#pragma unroll
    for (int j = 0; j < 4; ++j) {
      int nl = wn * 64 + j * 16 + (lane & 15);
      float bb = bias[wn * HH + slice + (nl & 63)];
#pragma unroll
      for (int i = 0; i < 2; ++i)
#pragma unroll
        for (int v = 0; v < 8; ++v) {
          int m = wm * 32 + i * 16 + v + ((lane >> 4) << 3);
          gsm[nl * 64 + m] = acc[i][j][v] + bb;
        }
    }
    __syncthreads();
    // combine gates -> c,h update (each thread owns 16 (b,j) pairs)
    float h2r[16], er[16];
#pragma unroll
    for (int it = 0; it < 16; ++it) {
      int idx = tid + it * 256;
      int b = idx & 63, j = idx >> 6;
      float iv = gsm[(0 * 64 + j) * 64 + b];
      float fv = gsm[(1 * 64 + j) * 64 + b];
      float gv = gsm[(2 * 64 + j) * 64 + b];
      float ov = gsm[(3 * 64 + j) * 64 + b];
      int col = slice + j;
      float c_old = cbuf[b * HH + col];
      float c_new = sigf(fv) * c_old + sigf(iv) * tanhf(gv);
      float h_new = sigf(ov) * tanhf(c_new);
      bool mk = ts < lengths[b];
      float h_old = (float)hbuf[b * HH + col];
      cbuf[b * HH + col] = mk ? c_new : c_old;
      h2r[it] = mk ? h_new : h_old;
      er[it] = mk ? h_new : 0.0f;   // layer output multiplied by mask
    }
    grid_barrier(sync);  // all WGs done reading h_t
#pragma unroll
    for (int it = 0; it < 16; ++it) {
      int idx = tid + it * 256;
      int b = idx & 63, j = idx >> 6;
      int col = slice + j;
      hbuf[b * HH + col] = (bf16_t)h2r[it];
      enc[((size_t)b * LL + ts) * (2 * HH) + dir * HH + col] = (bf16_t)er[it];
    }
    grid_barrier(sync);  // h_{t+1} input published
  }
}

// ---------------- small helper kernels ----------------
__global__ void sm_f2bf(const float* __restrict__ s, bf16_t* __restrict__ d, long n) {
  for (long i = (long)blockIdx.x * blockDim.x + threadIdx.x; i < n; i += (long)gridDim.x * blockDim.x)
    d[i] = (bf16_t)s[i];
}
__global__ void sm_wcat(const float* __restrict__ Wih, const float* __restrict__ Whh,
                        bf16_t* __restrict__ dst, int in_d) {
  const int K = HH + in_d;
  const long n = 2048L * K;
  for (long i = (long)blockIdx.x * blockDim.x + threadIdx.x; i < n; i += (long)gridDim.x * blockDim.x) {
    long r = i / K; int c = (int)(i - r * K);
    dst[i] = (bf16_t)(c < HH ? Whh[r * HH + c] : Wih[r * in_d + (c - HH)]);
  }
}
__global__ void sm_addb(const float* a, const float* b, float* d, int n) {
  int i = blockIdx.x * blockDim.x + threadIdx.x;
  if (i < n) d[i] = a[i] + b[i];
}
__global__ void sm_zero_bf(bf16_t* p, long n) {
  for (long i = (long)blockIdx.x * blockDim.x + threadIdx.x; i < n; i += (long)gridDim.x * blockDim.x) p[i] = (bf16_t)0.0f;
}
__global__ void sm_zero_f(float* p, long n) {
  for (long i = (long)blockIdx.x * blockDim.x + threadIdx.x; i < n; i += (long)gridDim.x * blockDim.x) p[i] = 0.0f;
}
__global__ void sm_zero_u(unsigned* p, int n) {
  int i = blockIdx.x * blockDim.x + threadIdx.x;
  if (i < n) p[i] = 0u;
}
__global__ void sm_embed(const int* __restrict__ ids, const float* __restrict__ emb,
                         bf16_t* __restrict__ x0, long n) {
  for (long i = (long)blockIdx.x * blockDim.x + threadIdx.x; i < n; i += (long)gridDim.x * blockDim.x) {
    long row = i >> 8; int e = (int)(i & 255);
    x0[i] = (bf16_t)emb[(size_t)ids[row] * EE + e];
  }
}
__global__ void sm_pooled(const bf16_t* __restrict__ enc, const int* __restrict__ lengths,
                          float* __restrict__ pooled, bf16_t* __restrict__ pbf) {
  int b = blockIdx.x;
  float inv = 1.0f / (float)lengths[b];
  for (int d = threadIdx.x; d < DD; d += 256) {
    float s = 0.0f;
    for (int l = 0; l < LL; ++l) s += (float)enc[((size_t)b * LL + l) * DD + d];
    float v = s * inv;
    pooled[b * DD + d] = v;
    pbf[b * DD + d] = (bf16_t)v;
  }
}
__global__ void sm_opemb(const float* __restrict__ op_emb, const int* __restrict__ targets,
                         bf16_t* __restrict__ gi, int t) {
  int b = blockIdx.x;
  int prev = (t == 0) ? START_OP : targets[b * TT + t - 1];
  gi[b * 1088 + threadIdx.x] = (bf16_t)op_emb[prev * OPE_D + threadIdx.x];
}
__global__ void sm_attn(const bf16_t* __restrict__ q, const bf16_t* __restrict__ enc,
                        const int* __restrict__ lengths,
                        float* __restrict__ ctx, bf16_t* __restrict__ gi) {
  __shared__ float sc[LL];
  __shared__ float red[256];
  int b = blockIdx.x, tid = threadIdx.x;
  int len = lengths[b];
  const bf16_t* qr = q + (size_t)b * DD;
  for (int l = tid; l < LL; l += 256) {
    const bf16_t* er = enc + ((size_t)b * LL + l) * DD;
    float a = 0.0f;
    for (int d = 0; d < DD; ++d) a += (float)qr[d] * (float)er[d];
    sc[l] = (l < len) ? a : NEGBIG;
  }
  __syncthreads();
  red[tid] = fmaxf(sc[tid], sc[tid + 256]);
  __syncthreads();
  for (int s = 128; s > 0; s >>= 1) { if (tid < s) red[tid] = fmaxf(red[tid], red[tid + s]); __syncthreads(); }
  float mx = red[0];
  __syncthreads();
  float sum = 0.0f;
  for (int l = tid; l < LL; l += 256) { float e = __expf(sc[l] - mx); sc[l] = e; sum += e; }
  red[tid] = sum;
  __syncthreads();
  for (int s = 128; s > 0; s >>= 1) { if (tid < s) red[tid] += red[tid + s]; __syncthreads(); }
  float inv = 1.0f / red[0];
  __syncthreads();
  for (int d = tid; d < DD; d += 256) {
    float a = 0.0f;
    for (int l = 0; l < LL; ++l) a += sc[l] * (float)enc[((size_t)b * LL + l) * DD + d];
    a *= inv;
    ctx[b * DD + d] = a;
    gi[b * 1088 + OPE_D + d] = (bf16_t)a;
  }
}
__global__ void sm_gru(const float* __restrict__ gx, const float* __restrict__ gh,
                       float* __restrict__ h, bf16_t* __restrict__ hbf) {
  int b = blockIdx.x;
  for (int d = threadIdx.x; d < DD; d += 256) {
    float rx = gx[b * 3072 + d],        rh = gh[b * 3072 + d];
    float zx = gx[b * 3072 + DD + d],   zh = gh[b * 3072 + DD + d];
    float nx = gx[b * 3072 + 2*DD + d], nh = gh[b * 3072 + 2*DD + d];
    float r = sigf(rx + rh), z = sigf(zx + zh);
    float n = tanhf(nx + r * nh);
    float hv = (1.0f - z) * n + z * h[b * DD + d];
    h[b * DD + d] = hv;
    hbf[b * DD + d] = (bf16_t)hv;
  }
}
__global__ void sm_span(const bf16_t* __restrict__ q, const bf16_t* __restrict__ enc,
                        const int* __restrict__ lengths, float* __restrict__ out) {
  int b = blockIdx.x, len = lengths[b];
  const bf16_t* qr = q + (size_t)b * DD;
  for (int l = threadIdx.x; l < LL; l += 256) {
    float s = NEGBIG;
    if (l < len) {
      const bf16_t* er = enc + ((size_t)b * LL + l) * DD;
      float a = 0.0f;
      for (int d = 0; d < DD; ++d) a += (float)qr[d] * (float)er[d];
      s = a;
    }
    out[(size_t)b * (TT * LL) + l] = s;
  }
}
__global__ void sm_ref(const float* __restrict__ q01, const float* __restrict__ k0,
                       const float* __restrict__ k1,
                       float* __restrict__ out0, float* __restrict__ out1, int t) {
  int b = blockIdx.x, tid = threadIdx.x;  // block 16: head = tid>>3, t' = tid&7
  int head = tid >> 3, tp = tid & 7;
  const float* q = q01 + (size_t)head * BB * RD_D + (size_t)b * RD_D;
  float v = NEGBIG;
  if (tp < t) {
    const float* kv = (head ? k1 : k0) + ((size_t)tp * BB + b) * RD_D;
    float a = 0.0f;
    for (int d = 0; d < RD_D; ++d) a += q[d] * kv[d];
    v = a;
  }
  (head ? out1 : out0)[(size_t)b * TT * TT + tp] = v;
}

// ---------------- host ----------------
extern "C" void kernel_launch(void* const* d_in, const int* in_sizes, int n_in,
                              void* d_out, int out_size, void* d_ws, size_t ws_size,
                              hipStream_t stream) {
  (void)in_sizes; (void)n_in; (void)out_size; (void)ws_size;
  const int*   ids     = (const int*)d_in[0];
  const int*   lengths = (const int*)d_in[1];
  const int*   targets = (const int*)d_in[2];
  const float* emb     = (const float*)d_in[3];
  float* out = (float*)d_out;

  char* base = (char*)d_ws;
  size_t off = 0;
  auto alloc = [&](size_t bytes) -> void* {
    off = (off + 255) & ~(size_t)255;
    void* p = base + off; off += bytes; return p;
  };
  bf16_t* x0   = (bf16_t*)alloc(8388608L * 2);           // (B,L,E)
  bf16_t* enc1 = (bf16_t*)alloc(33554432L * 2);          // (B,L,1024)
  bf16_t* enc2 = (bf16_t*)alloc(33554432L * 2);
  bf16_t* wc0f = (bf16_t*)alloc(2048L * 768 * 2);
  bf16_t* wc0b = (bf16_t*)alloc(2048L * 768 * 2);
  bf16_t* wc1f = (bf16_t*)alloc(2048L * 1536 * 2);
  bf16_t* wc1b = (bf16_t*)alloc(2048L * 1536 * 2);
  float*  bc0f = (float*)alloc(2048 * 4);
  float*  bc0b = (float*)alloc(2048 * 4);
  float*  bc1f = (float*)alloc(2048 * 4);
  float*  bc1b = (float*)alloc(2048 * 4);
  bf16_t* hst  = (bf16_t*)alloc(BB * HH * 2);
  float*  cst  = (float*)alloc(BB * HH * 4);
  unsigned* syn = (unsigned*)alloc(256);
  bf16_t* gWih = (bf16_t*)alloc(3072L * 1088 * 2);
  bf16_t* gWhh = (bf16_t*)alloc(3072L * 1024 * 2);
  bf16_t* ctxW = (bf16_t*)alloc(1024L * 1024 * 2);
  bf16_t* inhW = (bf16_t*)alloc(1024L * 1024 * 2);
  bf16_t* opW  = (bf16_t*)alloc(64L * 1024 * 2);
  bf16_t* a0W  = (bf16_t*)alloc(4L * 1024 * 2);
  bf16_t* a1W  = (bf16_t*)alloc(4L * 1024 * 2);
  bf16_t* spW[4];
  for (int i = 0; i < 4; ++i) spW[i] = (bf16_t*)alloc(1024L * 1024 * 2);
  bf16_t* rW[4];
  for (int i = 0; i < 4; ++i) rW[i] = (bf16_t*)alloc(128L * 1024 * 2);   // r0q, r0k, r1q, r1k
  bf16_t* litW0 = (bf16_t*)alloc(32000L * 1024 * 2);
  bf16_t* litW1 = (bf16_t*)alloc(32000L * 1024 * 2);
  float*  pooled = (float*)alloc(BB * DD * 4);
  bf16_t* pooledb = (bf16_t*)alloc(BB * DD * 2);
  float*  hf  = (float*)alloc(BB * DD * 4);
  bf16_t* hbf = (bf16_t*)alloc(BB * DD * 2);
  float*  qf  = (float*)alloc(BB * DD * 4);
  bf16_t* qbf = (bf16_t*)alloc(BB * DD * 2);
  bf16_t* gi  = (bf16_t*)alloc(BB * 1088 * 2);
  float*  ctxf = (float*)alloc(BB * DD * 4);
  float*  gx  = (float*)alloc(BB * 3072 * 4);
  float*  gh  = (float*)alloc(BB * 3072 * 4);
  float*  kb0 = (float*)alloc((size_t)TT * BB * RD_D * 4);
  float*  kb1 = (float*)alloc((size_t)TT * BB * RD_D * 4);
  float*  qr  = (float*)alloc(2L * BB * RD_D * 4);
  bf16_t* spq = (bf16_t*)alloc(BB * DD * 2);

  auto cvt = [&](int idx, bf16_t* d, long n) {
    sm_f2bf<<<1024, 256, 0, stream>>>((const float*)d_in[idx], d, n);
  };
  // ---- weight preparation (bf16, L2-resident working set) ----
  sm_zero_u<<<1, 64, 0, stream>>>(syn, 64);
  sm_wcat<<<2048, 256, 0, stream>>>((const float*)d_in[4],  (const float*)d_in[5],  wc0f, EE);
  sm_wcat<<<2048, 256, 0, stream>>>((const float*)d_in[8],  (const float*)d_in[9],  wc0b, EE);
  sm_wcat<<<2048, 256, 0, stream>>>((const float*)d_in[12], (const float*)d_in[13], wc1f, 2 * HH);
  sm_wcat<<<2048, 256, 0, stream>>>((const float*)d_in[16], (const float*)d_in[17], wc1b, 2 * HH);
  sm_addb<<<8, 256, 0, stream>>>((const float*)d_in[6],  (const float*)d_in[7],  bc0f, 2048);
  sm_addb<<<8, 256, 0, stream>>>((const float*)d_in[10], (const float*)d_in[11], bc0b, 2048);
  sm_addb<<<8, 256, 0, stream>>>((const float*)d_in[14], (const float*)d_in[15], bc1f, 2048);
  sm_addb<<<8, 256, 0, stream>>>((const float*)d_in[18], (const float*)d_in[19], bc1b, 2048);
  cvt(25, gWih, 3072L * 1088);
  cvt(26, gWhh, 3072L * 1024);
  cvt(23, ctxW, 1024L * 1024);
  cvt(21, inhW, 1024L * 1024);
  cvt(29, opW, 64L * 1024);
  cvt(31, a0W, 4L * 1024);
  cvt(33, a1W, 4L * 1024);
  { int wi[4] = {35, 37, 39, 41}; for (int i = 0; i < 4; ++i) cvt(wi[i], spW[i], 1024L * 1024); }
  { int wi[4] = {43, 45, 47, 49}; for (int i = 0; i < 4; ++i) cvt(wi[i], rW[i], 128L * 1024); }
  cvt(51, litW0, 32000L * 1024);
  cvt(53, litW1, 32000L * 1024);

  // ---- encoder ----
  sm_embed<<<4096, 256, 0, stream>>>(ids, emb, x0, 8388608L);
  struct { bf16_t* w; float* b; bf16_t* xin; bf16_t* eo; int in_d; int rev; int dir; } disp[4] = {
    {wc0f, bc0f, x0,   enc1, EE,     0, 0},
    {wc0b, bc0b, x0,   enc1, EE,     1, 1},
    {wc1f, bc1f, enc1, enc2, 2 * HH, 0, 0},
    {wc1b, bc1b, enc1, enc2, 2 * HH, 1, 1},
  };
  for (int i = 0; i < 4; ++i) {
    sm_zero_bf<<<64, 256, 0, stream>>>(hst, (long)BB * HH);
    sm_zero_f<<<64, 256, 0, stream>>>(cst, (long)BB * HH);
    sm_lstm_persist<<<NBLK, 256, 0, stream>>>(disp[i].w, disp[i].b, disp[i].xin, lengths,
                                              hst, cst, disp[i].eo, syn,
                                              disp[i].in_d, disp[i].rev, disp[i].dir);
  }

  // ---- decoder init ----
  sm_pooled<<<BB, 256, 0, stream>>>(enc2, lengths, pooled, pooledb);
  sm_gemm_bf16<<<dim3(8, 1), 256, 0, stream>>>(pooledb, DD, inhW, DD, (const float*)d_in[22],
                                               hf, DD, hbf, DD, BB, DD, DD, /*tanh*/1);

  // d_out float offsets
  const size_t OP_O = 0, A0_O = 32768, A1_O = 34816;
  const size_t SP_O[4] = {36864, 299008, 561152, 823296};
  const size_t R0_O = 1085440, R1_O = 1089536, L0_O = 1093632, L1_O = 17477632;
  const int spb[4] = {36, 38, 40, 42};

  for (int t = 0; t < TT; ++t) {
    sm_opemb<<<BB, 64, 0, stream>>>((const float*)d_in[20], targets, gi, t);
    sm_gemm_bf16<<<dim3(8, 1), 256, 0, stream>>>(hbf, DD, ctxW, DD, (const float*)d_in[24],
                                                 qf, DD, qbf, DD, BB, DD, DD, 0);
    sm_attn<<<BB, 256, 0, stream>>>(qbf, enc2, lengths, ctxf, gi);
    sm_gemm_bf16<<<dim3(24, 1), 256, 0, stream>>>(gi, 1088, gWih, 1088, (const float*)d_in[27],
                                                  gx, 3072, nullptr, 0, BB, 3072, 1088, 0);
    sm_gemm_bf16<<<dim3(24, 1), 256, 0, stream>>>(hbf, DD, gWhh, DD, (const float*)d_in[28],
                                                  gh, 3072, nullptr, 0, BB, 3072, DD, 0);
    sm_gru<<<BB, 256, 0, stream>>>(gx, gh, hf, hbf);
    // ref key/query projections
    sm_gemm_bf16<<<dim3(1, 1), 256, 0, stream>>>(hbf, DD, rW[1], DD, (const float*)d_in[46],
                                                 kb0 + (size_t)t * BB * RD_D, RD_D, nullptr, 0, BB, RD_D, DD, 0);
    sm_gemm_bf16<<<dim3(1, 1), 256, 0, stream>>>(hbf, DD, rW[3], DD, (const float*)d_in[50],
                                                 kb1 + (size_t)t * BB * RD_D, RD_D, nullptr, 0, BB, RD_D, DD, 0);
    sm_gemm_bf16<<<dim3(1, 1), 256, 0, stream>>>(hbf, DD, rW[0], DD, (const float*)d_in[44],
                                                 qr, RD_D, nullptr, 0, BB, RD_D, DD, 0);
    sm_gemm_bf16<<<dim3(1, 1), 256, 0, stream>>>(hbf, DD, rW[2], DD, (const float*)d_in[48],
                                                 qr + (size_t)BB * RD_D, RD_D, nullptr, 0, BB, RD_D, DD, 0);
    sm_ref<<<BB, 16, 0, stream>>>(qr, kb0, kb1, out + R0_O + t * TT, out + R1_O + t * TT, t);
    // small heads, written straight into d_out slices
    sm_gemm_bf16<<<dim3(1, 1), 256, 0, stream>>>(hbf, DD, opW, DD, (const float*)d_in[30],
                                                 out + OP_O + t * 64, TT * 64, nullptr, 0, BB, 64, DD, 0);
    sm_gemm_bf16<<<dim3(1, 1), 256, 0, stream>>>(hbf, DD, a0W, DD, (const float*)d_in[32],
                                                 out + A0_O + t * 4, TT * 4, nullptr, 0, BB, 4, DD, 0);
    sm_gemm_bf16<<<dim3(1, 1), 256, 0, stream>>>(hbf, DD, a1W, DD, (const float*)d_in[34],
                                                 out + A1_O + t * 4, TT * 4, nullptr, 0, BB, 4, DD, 0);
    // span heads
    for (int s = 0; s < 4; ++s) {
      sm_gemm_bf16<<<dim3(8, 1), 256, 0, stream>>>(hbf, DD, spW[s], DD, (const float*)d_in[spb[s]],
                                                   nullptr, 0, spq, DD, BB, DD, DD, 0);
      sm_span<<<BB, 256, 0, stream>>>(spq, enc2, lengths, out + SP_O[s] + (size_t)t * LL);
    }
    // vocab heads: bf16 weights are L2-resident across all 8 steps
    sm_gemm_bf16<<<dim3(250, 1), 256, 0, stream>>>(hbf, DD, litW0, DD, (const float*)d_in[52],
                                                   out + L0_O + (size_t)t * VOCAB, TT * VOCAB,
                                                   nullptr, 0, BB, VOCAB, DD, 0);
    sm_gemm_bf16<<<dim3(250, 1), 256, 0, stream>>>(hbf, DD, litW1, DD, (const float*)d_in[54],
                                                   out + L1_O + (size_t)t * VOCAB, TT * VOCAB,
                                                   nullptr, 0, BB, VOCAB, DD, 0);
  }
}
